// EMAVectorQuantizer_12970801234462
// MI455X (gfx1250) — matrix operand order repaired and checked by reading, not compiled, not run on hardware
//
#include <hip/hip_runtime.h>
#include <hip/hip_bf16.h>
#include <math.h>

// ---------------------------------------------------------------------------
// EMA Vector Quantizer for MI455X (gfx1250, wave32, WMMA, async-to-LDS).
//
// Hot loop: distances = ||e||^2 - 2 * x . e over [32768 x 8192], argmin over K.
// bf16 split-precision WMMA (hi/lo decomposition, f32 accumulate):
//   x.w ~= hiA*hiB + hiA*loB + loA*hiB   (3x v_wmma_f32_16x16x32_bf16)
// Codebook chunks are staged global->LDS with async copies (double-buffered),
// shared by all 8 waves of the workgroup, consumed via ds_load_b128.
// Working set is L2-resident (192 MB L2) -> pure matrix-pipe bound.
// ---------------------------------------------------------------------------

typedef __attribute__((ext_vector_type(16))) __bf16 v16bf;
typedef __attribute__((ext_vector_type(8)))  __bf16 v8bf;
typedef __attribute__((ext_vector_type(8)))  float  v8f;

// exact parameter types of the async-to-LDS builtin (from hipcc diagnostics):
// arg0: int __vector(4) in global AS, arg1: LDS destination
typedef int v4i_vs __attribute__((__vector_size__(16)));
typedef __attribute__((address_space(1))) v4i_vs AS1v4i;
typedef __attribute__((address_space(3))) v4i_vs AS3v4i;

#define NUM_EMB 8192
#define DIM     64
#define NROWS   32768          // 8*4096
#define NELEM   (NROWS * DIM)  // 2097152
#define DECAY   0.99f
#define ONE_M_DECAY 0.01f
#define EPSV    1e-5f
#define COMMIT  0.25f

#define CHUNK   64                    // codebook rows staged per step
#define NCHUNK  (NUM_EMB / CHUNK)     // 128
#define LROW    72                    // padded LDS row stride in bf16 (144 B)

// Async global->LDS path, guarded so the file compiles on any toolchain.
#if defined(__has_builtin)
#if __has_builtin(__builtin_amdgcn_global_load_async_to_lds_b128) && \
    __has_builtin(__builtin_amdgcn_s_wait_asynccnt)
#define USE_ASYNC 1
#endif
#endif
#ifndef USE_ASYNC
#define USE_ASYNC 0
#endif

static __device__ inline v8f wmma_bf16(v16bf a, v16bf b, v8f c) {
  return __builtin_amdgcn_wmma_f32_16x16x32_bf16(
      /*neg_a=*/false, a, /*neg_b=*/false, b,
      /*c_mod=*/(short)0, c, /*reuse_a=*/false, /*reuse_b=*/false);
}

// ---------------------------------------------------------------------------
// 0) zero accumulator region of workspace
// ---------------------------------------------------------------------------
__global__ void vq_zero_kernel(float* p, int n) {
  int t = blockIdx.x * blockDim.x + threadIdx.x;
  if (t < n) p[t] = 0.0f;
}

// ---------------------------------------------------------------------------
// 1) prep: split codebook to bf16 hi/lo, compute ||e_k||^2
// ---------------------------------------------------------------------------
__global__ void vq_prep_kernel(const float* __restrict__ weight,
                               __bf16* __restrict__ whi,
                               __bf16* __restrict__ wlo,
                               float* __restrict__ wsq) {
  int k = blockIdx.x * blockDim.x + threadIdx.x;
  if (k >= NUM_EMB) return;
  float s = 0.0f;
  const float* wr = weight + k * DIM;
  __bf16* hr = whi + k * DIM;
  __bf16* lr = wlo + k * DIM;
  for (int d = 0; d < DIM; ++d) {
    float w = wr[d];
    s += w * w;
    __bf16 h = (__bf16)w;
    hr[d] = h;
    lr[d] = (__bf16)(w - (float)h);
  }
  wsq[k] = s;
}

// ---------------------------------------------------------------------------
// stage one 64-row codebook chunk (hi+lo) into padded LDS buffers.
// 256 threads x 16B x 2 sweeps per matrix.  Row stride 144B in LDS
// (36 dword-banks, gcd(36,64)=4 -> conflict-free half-wave fragment reads).
// ---------------------------------------------------------------------------
static __device__ inline void stage_chunk(const __bf16* __restrict__ whi,
                                          const __bf16* __restrict__ wlo,
                                          __bf16* dst_hi, __bf16* dst_lo,
                                          int chunk, int tid) {
  const char* gh = (const char*)(whi + (size_t)chunk * CHUNK * DIM);
  const char* gl = (const char*)(wlo + (size_t)chunk * CHUNK * DIM);
  #pragma unroll
  for (int sweep = 0; sweep < 2; ++sweep) {
    int o = (tid + sweep * 256) * 16;   // byte offset in 8KB chunk image
    int row = o >> 7;                   // /128 (row = 128B in global)
    int inrow = o & 127;                // 16B aligned
    char* lh = (char*)dst_hi + row * (LROW * 2) + inrow;
    char* ll = (char*)dst_lo + row * (LROW * 2) + inrow;
#if USE_ASYNC
    __builtin_amdgcn_global_load_async_to_lds_b128(
        (AS1v4i*)(gh + o), (AS3v4i*)lh, 0, 0);
    __builtin_amdgcn_global_load_async_to_lds_b128(
        (AS1v4i*)(gl + o), (AS3v4i*)ll, 0, 0);
#else
    *(float4*)lh = *(const float4*)(gh + o);
    *(float4*)ll = *(const float4*)(gl + o);
#endif
  }
}

static __device__ inline v16bf load_frag_lds(const __bf16* p) {
  union { v16bf v; v8bf h[2]; } u;
  u.h[0] = *(const v8bf*)(p);        // 16B, ds_load_b128
  u.h[1] = *(const v8bf*)(p + 8);    // 16B, ds_load_b128
  return u.v;
}

// ---------------------------------------------------------------------------
// 2) main fused GEMM + argmin.  One wave per 16-row strip (8 waves / block,
//    128 rows / block, 256 blocks).  Codebook double-buffered through LDS.
//    EXEC is all-ones throughout (no divergence around WMMA).
// ---------------------------------------------------------------------------
__global__ void __launch_bounds__(256)
vq_argmin_kernel(const float*  __restrict__ flat,   // [NROWS, DIM]
                 const __bf16* __restrict__ whi,    // [NUM_EMB, DIM]
                 const __bf16* __restrict__ wlo,    // [NUM_EMB, DIM]
                 const float*  __restrict__ wsq,    // [NUM_EMB]
                 int*          __restrict__ idx_out,
                 float*        __restrict__ idx_f_out) {
  __shared__ __bf16 lds_hi[2][CHUNK * LROW];   // 2 x 9 KB
  __shared__ __bf16 lds_lo[2][CHUNK * LROW];   // 2 x 9 KB   (36 KB total)

  const int tid   = threadIdx.x;
  const int lane  = tid & 31;
  const int rlane = lane & 15;       // column within 16-wide tile / row-in-strip
  const int hi16  = lane >> 4;       // upper half-wave flag
  const int wave  = blockIdx.x * (blockDim.x >> 5) + (tid >> 5);
  const int rowBase = wave * 16;

  // ---- A fragments: 16x32 bf16, loaded once, converted f32 -> hi/lo bf16.
  // Layout (contiguous k, matching B): lane L(<16)=row L holds K 0..15;
  // lane L+16 holds K 16..31.  Two K-steps cover D=64.
  v16bf a_hi[2], a_lo[2];
  const float* xrow = flat + (size_t)(rowBase + rlane) * DIM;
  #pragma unroll
  for (int s = 0; s < 2; ++s) {
    const int kb0 = 32 * s + hi16 * 16;       // elements 0..7
    const int kb1 = 32 * s + hi16 * 16 + 8;   // elements 8..15
    #pragma unroll
    for (int e = 0; e < 8; ++e) {
      float x0 = xrow[kb0 + e];
      __bf16 h0 = (__bf16)x0;
      a_hi[s][e] = h0;
      a_lo[s][e] = (__bf16)(x0 - (float)h0);
      float x1 = xrow[kb1 + e];
      __bf16 h1 = (__bf16)x1;
      a_hi[s][8 + e] = h1;
      a_lo[s][8 + e] = (__bf16)(x1 - (float)h1);
    }
  }

  float best[8];
  int   bidx[8];
  #pragma unroll
  for (int r = 0; r < 8; ++r) { best[r] = 3.4e38f; bidx[r] = 0; }

  // prologue: stage chunk 0 into buffer 0
  stage_chunk(whi, wlo, lds_hi[0], lds_lo[0], 0, tid);

  for (int c = 0; c < NCHUNK; ++c) {
    const int buf = c & 1;
#if USE_ASYNC
    __builtin_amdgcn_s_wait_asynccnt(0);   // my staged portion has landed
#endif
    __syncthreads();                       // everyone's portion has landed;
                                           // also orders compute(c-1) before
                                           // stage(c+1) overwrite below
    if (c + 1 < NCHUNK)
      stage_chunk(whi, wlo, lds_hi[buf ^ 1], lds_lo[buf ^ 1], c + 1, tid);
    if (c + 2 < NCHUNK)                    // warm GL2 for the chunk after next
      __builtin_prefetch(whi + (size_t)(c + 2) * CHUNK * DIM + tid * 16, 0, 0);

    // ---- 4 column-tiles per resident chunk
    #pragma unroll
    for (int t = 0; t < 4; ++t) {
      const int lrow = t * 16 + rlane;          // row within chunk
      const int col  = c * CHUNK + lrow;        // global codebook index
      // B fragment (32x16 bf16): lane L(<16)=col L holds K 0..15; lane L+16
      // holds K 16..31 -> 16 consecutive bf16 at dim offset hi16*16 (+32).
      const __bf16* bh = &lds_hi[buf][lrow * LROW + hi16 * 16];
      const __bf16* bl = &lds_lo[buf][lrow * LROW + hi16 * 16];
      v16bf bh0 = load_frag_lds(bh);
      v16bf bl0 = load_frag_lds(bl);
      v16bf bh1 = load_frag_lds(bh + 32);
      v16bf bl1 = load_frag_lds(bl + 32);

      v8f acc = {};
      acc = wmma_bf16(a_hi[0], bh0, acc);   // hi*hi (k-step 0)
      acc = wmma_bf16(a_hi[0], bl0, acc);   // hi*lo
      acc = wmma_bf16(a_lo[0], bh0, acc);   // lo*hi
      acc = wmma_bf16(a_hi[1], bh1, acc);   // k-step 1
      acc = wmma_bf16(a_hi[1], bl1, acc);
      acc = wmma_bf16(a_lo[1], bh1, acc);

      const float wc = wsq[col];
      #pragma unroll
      for (int r = 0; r < 8; ++r) {
        float d = wc - 2.0f * acc[r];   // ||e||^2 - 2 x.e (||x||^2 drops out)
        if (d < best[r]) { best[r] = d; bidx[r] = col; }  // first idx on ties
      }
    }
  }

  // ---- per-row argmin across the 16 column-lanes (masks <= 8 keep the
  // butterfly inside each 16-lane half)
  #pragma unroll
  for (int r = 0; r < 8; ++r) {
    float bv = best[r];
    int   bi = bidx[r];
    #pragma unroll
    for (int m = 8; m >= 1; m >>= 1) {
      float ov = __shfl_xor(bv, m, 32);
      int   oi = __shfl_xor(bi, m, 32);
      if (ov < bv || (ov == bv && oi < bi)) { bv = ov; bi = oi; }
    }
    if (rlane == 0) {
      int row = rowBase + hi16 * 8 + r;
      idx_out[row]   = bi;
      idx_f_out[row] = (float)bi;   // idx emitted as float into the f32 d_out
    }
  }
}

// ---------------------------------------------------------------------------
// 3) segment sums: counts[k] and emb_sum[k][d] via L2 atomics
// ---------------------------------------------------------------------------
__global__ void vq_stats_kernel(const float* __restrict__ flat,
                                const int* __restrict__ idx,
                                float* __restrict__ counts,
                                float* __restrict__ emb_sum) {
  int t = blockIdx.x * blockDim.x + threadIdx.x;
  if (t >= NELEM) return;
  int n = t >> 6, d = t & 63;
  int k = idx[n];
  atomicAdd(&emb_sum[k * DIM + d], flat[t]);
  if (d == 0) atomicAdd(&counts[k], 1.0f);
}

// ---------------------------------------------------------------------------
// 4) new_cc, n = sum(new_cc), entropy accumulation (wave-reduced atomics)
// ---------------------------------------------------------------------------
__global__ void vq_cc_kernel(const float* __restrict__ ema_cc,
                             const float* __restrict__ counts,
                             float* __restrict__ new_cc_out,
                             float* __restrict__ scal) {
  int k = blockIdx.x * blockDim.x + threadIdx.x;
  float nc = 0.0f, ent = 0.0f;
  if (k < NUM_EMB) {
    nc = ema_cc[k] * DECAY + ONE_M_DECAY * counts[k];
    new_cc_out[k] = nc;
    float p = counts[k] * (1.0f / (float)NROWS);
    ent = -p * logf(p + 1e-10f);
  }
  #pragma unroll
  for (int m = 16; m >= 1; m >>= 1) {
    nc  += __shfl_xor(nc,  m, 32);
    ent += __shfl_xor(ent, m, 32);
  }
  if ((threadIdx.x & 31) == 0) {
    atomicAdd(&scal[0], nc);    // n
    atomicAdd(&scal[1], ent);   // -sum p log p
  }
}

// ---------------------------------------------------------------------------
// 5) new_ws, smoothed counts, new_weight
// ---------------------------------------------------------------------------
__global__ void vq_weight_kernel(const float* __restrict__ ema_ws,
                                 const float* __restrict__ emb_sum,
                                 const float* __restrict__ new_cc,
                                 const float* __restrict__ scal,
                                 float* __restrict__ new_ws_out,
                                 float* __restrict__ new_w_out) {
  int t = blockIdx.x * blockDim.x + threadIdx.x;
  if (t >= NUM_EMB * DIM) return;
  int k = t >> 6;
  float nws = ema_ws[t] * DECAY + ONE_M_DECAY * emb_sum[t];
  new_ws_out[t] = nws;
  float n = scal[0];
  float smoothed = (new_cc[k] + EPSV) / (n + (float)NUM_EMB * EPSV) * n;
  new_w_out[t] = nws / smoothed;
}

// ---------------------------------------------------------------------------
// 6) quantized = new_weight[idx]; accumulate SSE (block-reduced atomic)
// ---------------------------------------------------------------------------
__global__ void __launch_bounds__(256)
vq_quant_kernel(const float* __restrict__ flat,
                const int* __restrict__ idx,
                const float* __restrict__ new_w,
                float* __restrict__ quant_out,
                float* __restrict__ scal) {
  __shared__ float sred[8];
  int t = blockIdx.x * 256 + threadIdx.x;
  float df = 0.0f;
  if (t < NELEM) {
    int n = t >> 6, d = t & 63;
    float q = new_w[idx[n] * DIM + d];
    quant_out[t] = q;                 // straight-through == quantized_raw value
    float e = q - flat[t];
    df = e * e;
  }
  #pragma unroll
  for (int m = 16; m >= 1; m >>= 1) df += __shfl_xor(df, m, 32);
  if ((threadIdx.x & 31) == 0) sred[threadIdx.x >> 5] = df;
  __syncthreads();
  if (threadIdx.x < 8) {
    float v = sred[threadIdx.x];
    #pragma unroll
    for (int m = 4; m >= 1; m >>= 1) v += __shfl_xor(v, m, 32);
    if (threadIdx.x == 0) atomicAdd(&scal[2], v);
  }
}

// ---------------------------------------------------------------------------
// 7) scalar outputs
// ---------------------------------------------------------------------------
__global__ void vq_final_kernel(const float* __restrict__ scal,
                                float* __restrict__ vq_loss_out,
                                float* __restrict__ perp_out) {
  vq_loss_out[0] = COMMIT * scal[2] * (1.0f / (float)NELEM);
  perp_out[0]    = expf(scal[1]);
}

// ---------------------------------------------------------------------------
extern "C" void kernel_launch(void* const* d_in, const int* in_sizes, int n_in,
                              void* d_out, int out_size, void* d_ws, size_t ws_size,
                              hipStream_t stream) {
  const float* flat   = (const float*)d_in[0];   // [8,4096,64] -> [32768,64]
  const float* weight = (const float*)d_in[1];   // [8192,64]
  const float* ema_cc = (const float*)d_in[2];   // [8192]
  const float* ema_ws = (const float*)d_in[3];   // [8192,64]

  // ---- d_out layout (tuple, flat, return order) ----
  float* out_quant  = (float*)d_out;                       // 2097152
  float* out_vqloss = out_quant  + NELEM;                  // 1
  float* out_idx    = out_vqloss + 1;                      // 32768 (as float)
  float* out_perp   = out_idx    + NROWS;                  // 1
  float* out_neww   = out_perp   + 1;                      // 524288
  float* out_newcc  = out_neww   + NUM_EMB * DIM;          // 8192
  float* out_newws  = out_newcc  + NUM_EMB;                // 524288

  // ---- workspace layout ----
  char* ws = (char*)d_ws;
  float*  wsq     = (float*)(ws);                                  // 32 KB
  __bf16* whi     = (__bf16*)(ws + 32768);                         // 1 MB
  __bf16* wlo     = (__bf16*)(ws + 32768 + 1048576);               // 1 MB
  float*  counts  = (float*)(ws + 2129920);                        // 32 KB
  float*  emb_sum = (float*)(ws + 2162688);                        // 2 MB
  int*    idx     = (int*)  (ws + 4259840);                        // 128 KB
  float*  scal    = (float*)(ws + 4390912);                        // n, ent, sse

  // 0) zero accumulators (counts + emb_sum + idx + scal)
  {
    int nz = (4390912 + 32 - 2129920) / 4;   // 565256 floats
    vq_zero_kernel<<<(nz + 255) / 256, 256, 0, stream>>>(counts, nz);
  }
  // 1) codebook prep
  vq_prep_kernel<<<(NUM_EMB + 255) / 256, 256, 0, stream>>>(weight, whi, wlo, wsq);
  // 2) fused distance GEMM + argmin: 2048 waves, 8 waves/block
  vq_argmin_kernel<<<256, 256, 0, stream>>>(flat, whi, wlo, wsq, idx, out_idx);
  // 3) segment sums
  vq_stats_kernel<<<(NELEM + 255) / 256, 256, 0, stream>>>(flat, idx, counts, emb_sum);
  // 4) new_cc + scalar reductions
  vq_cc_kernel<<<(NUM_EMB + 255) / 256, 256, 0, stream>>>(ema_cc, counts, out_newcc, scal);
  // 5) new_ws + new_weight
  vq_weight_kernel<<<(NUM_EMB * DIM + 255) / 256, 256, 0, stream>>>(
      ema_ws, emb_sum, out_newcc, scal, out_newws, out_neww);
  // 6) quantized gather + SSE
  vq_quant_kernel<<<(NELEM + 255) / 256, 256, 0, stream>>>(flat, idx, out_neww, out_quant, scal);
  // 7) scalars
  vq_final_kernel<<<1, 1, 0, stream>>>(scal, out_vqloss, out_perp);
}
